// Qwen2MoeSparseMoeBlock_69398081569502
// MI455X (gfx1250) — matrix operand order repaired and checked by reading, not compiled, not run on hardware
//
#include <hip/hip_runtime.h>

// ---------------------------------------------------------------------------
// Qwen2-MoE sparse block for MI455X (gfx1250, wave32, WMMA).
// Memory-bound: ~2.2GB f32 weights @ 23.3TB/s ~= 100us floor. Weights are
// converted f32->bf16 on the fly into LDS (software-pipelined so HBM latency
// overlaps WMMA); tokens gathered per expert (router top-4); bf16 WMMA
// (V_WMMA_F32_16X16X32_BF16) with f32 accum. A-tiles (already bf16) move
// global->LDS via GLOBAL_LOAD_ASYNC_TO_LDS_B128 (ASYNCcnt-tracked).
// ---------------------------------------------------------------------------

#define T_TOK 2048
#define HDIM  2048
#define FDIM  1408
#define NEXP  60
#define TOPK  4
#define SFDIM 5632

typedef __bf16 bf16x4  __attribute__((ext_vector_type(4)));
typedef __bf16 bf16x8  __attribute__((ext_vector_type(8)));
typedef __bf16 bf16x16 __attribute__((ext_vector_type(16)));
typedef float  v8f     __attribute__((ext_vector_type(8)));
typedef int    v4i     __attribute__((ext_vector_type(4)));

#define TILE_M 64
#define TILE_N 128
#define TILE_K 32
#define LDK    40   // padded LDS row stride (elements); 80B rows, 16B aligned,
                    // 20-bank step => conflict-free frag loads & b128 stores

// ---------------------------------------------------------------------------
// CDNA5 async global->LDS copy (ASYNCcnt-tracked)
// ---------------------------------------------------------------------------
#if __has_builtin(__builtin_amdgcn_global_load_async_to_lds_b128)
#define HAVE_ASYNC_LDS 1
#else
#define HAVE_ASYNC_LDS 0
#endif

__device__ __forceinline__ void async_cp16(void* lds_dst, const void* g_src) {
#if HAVE_ASYNC_LDS
  __builtin_amdgcn_global_load_async_to_lds_b128(
      (__attribute__((address_space(1))) v4i*)g_src,
      (__attribute__((address_space(3))) v4i*)lds_dst, 0, 0);
#else
  *reinterpret_cast<uint4*>(lds_dst) = *reinterpret_cast<const uint4*>(g_src);
#endif
}

__device__ __forceinline__ void async_wait0() {
#if HAVE_ASYNC_LDS
#if __has_builtin(__builtin_amdgcn_s_wait_asynccnt)
  __builtin_amdgcn_s_wait_asynccnt(0);
#else
  asm volatile("s_wait_asynccnt 0" ::: "memory");
#endif
#endif
}

// ---------------------------------------------------------------------------
// WMMA helpers
// ---------------------------------------------------------------------------
__device__ __forceinline__ v8f wmma_bf16(bf16x16 a, bf16x16 b, v8f c) {
  return __builtin_amdgcn_wmma_f32_16x16x32_bf16(false, a, false, b,
                                                 (short)0, c, false, false);
}

// A fragment (16x32 bf16, MxK): lane l<16 -> row M=l, K 0..15;
// lane l>=16 -> row M=l-16, K 16..31. Tile row-major [., LDK].
__device__ __forceinline__ bf16x16 ld_frag_a(const __bf16* tile, int m0, int lane) {
  int half = lane >> 4;
  int m    = m0 + (lane & 15);
  const __bf16* p = tile + m * LDK + half * 16;
  bf16x16 f;
  *reinterpret_cast<bf16x8*>(&f)       = *reinterpret_cast<const bf16x8*>(p);
  *(reinterpret_cast<bf16x8*>(&f) + 1) = *reinterpret_cast<const bf16x8*>(p + 8);
  return f;
}

// B fragment (32x16 bf16, KxN) stored transposed Bt[n][k]:
// lane l<16 -> col N=l, K 0..15; lane l>=16 -> col N=l-16, K 16..31.
__device__ __forceinline__ bf16x16 ld_frag_b(const __bf16* tileT, int n0, int lane) {
  int half = lane >> 4;
  int n    = n0 + (lane & 15);
  const __bf16* p = tileT + n * LDK + half * 16;
  bf16x16 f;
  *reinterpret_cast<bf16x8*>(&f)       = *reinterpret_cast<const bf16x8*>(p);
  *(reinterpret_cast<bf16x8*>(&f) + 1) = *reinterpret_cast<const bf16x8*>(p + 8);
  return f;
}

// ---------------------------------------------------------------------------
// Small prep kernels
// ---------------------------------------------------------------------------
__global__ void zero_counts_kernel(int* counts) {
  if (threadIdx.x < NEXP) counts[threadIdx.x] = 0;
}

__global__ __launch_bounds__(64)
void router_kernel(const float* __restrict__ x, const float* __restrict__ rw,
                   const float* __restrict__ sheg,
                   int* __restrict__ topidx, float* __restrict__ topw,
                   float* __restrict__ shgate, int* __restrict__ counts) {
  int t = blockIdx.x, tid = threadIdx.x;
  __shared__ float logits[64];
  __shared__ float red[64];
  const float* xt = x + (size_t)t * HDIM;

  float acc = 0.f;
  if (tid < NEXP) {
    for (int h = 0; h < HDIM; ++h) acc += xt[h] * rw[(size_t)h * NEXP + tid];
  }
  logits[tid] = (tid < NEXP) ? acc : -INFINITY;

  float g = 0.f;
  for (int h = tid; h < HDIM; h += 64) g += xt[h] * sheg[h];
  red[tid] = g;
  __syncthreads();

  if (tid == 0) {
    float s = 0.f;
    for (int i = 0; i < 64; ++i) s += red[i];
    shgate[t] = 1.f / (1.f + __expf(-s));

    float mx = -INFINITY;
    for (int e = 0; e < NEXP; ++e) mx = fmaxf(mx, logits[e]);
    float sum = 0.f;
    for (int e = 0; e < NEXP; ++e) {
      float p = __expf(logits[e] - mx);
      logits[e] = p;
      sum += p;
    }
    float inv = 1.f / sum;
    for (int k = 0; k < TOPK; ++k) {
      int bi = 0; float bv = -1.f;
      for (int e = 0; e < NEXP; ++e)
        if (logits[e] > bv) { bv = logits[e]; bi = e; }
      logits[bi] = -1.f;
      topidx[t * TOPK + k] = bi;
      topw[t * TOPK + k]   = bv * inv;   // NORM_TOPK == False
      atomicAdd(&counts[bi], 1);
    }
  }
}

__global__ void prefix_kernel(const int* __restrict__ counts,
                              int* __restrict__ offsets, int* __restrict__ cursors) {
  if (threadIdx.x == 0) {
    int s = 0;
    for (int e = 0; e < NEXP; ++e) { offsets[e] = s; s += counts[e]; }
    offsets[NEXP] = s;   // == T_TOK*TOPK
  }
  if (threadIdx.x < NEXP) cursors[threadIdx.x] = 0;
}

__global__ __launch_bounds__(256)
void scatter_kernel(const int* __restrict__ topidx, const float* __restrict__ topw,
                    const int* __restrict__ offsets, int* __restrict__ cursors,
                    int* __restrict__ tok, float* __restrict__ wgt) {
  int t = blockIdx.x * blockDim.x + threadIdx.x;
  if (t >= T_TOK) return;
  for (int k = 0; k < TOPK; ++k) {
    int e    = topidx[t * TOPK + k];
    int pos  = atomicAdd(&cursors[e], 1);
    int slot = offsets[e] + pos;
    tok[slot] = t;
    wgt[slot] = topw[t * TOPK + k];
  }
}

__global__ __launch_bounds__(256)
void cvt_x_kernel(const float* __restrict__ x, __bf16* __restrict__ xb) {
  size_t i = ((size_t)blockIdx.x * blockDim.x + threadIdx.x) * 4;
  float4 v = *reinterpret_cast<const float4*>(x + i);
  bf16x4 b = { (__bf16)v.x, (__bf16)v.y, (__bf16)v.z, (__bf16)v.w };
  *reinterpret_cast<bf16x4*>(xb + i) = b;
}

// ---------------------------------------------------------------------------
// Stage 1: Out[slot, n] = bf16( silu(x@Wg) * (x@Wu) )
//   64(M) x 128(N) block tile, K-step 32 over HDIM; 8 waves = 2(M) x 4(N).
//   N is a template constant: strided weight loads use immediate offsets.
//   B path is software-pipelined (regs for iter i+1 fetched during iter i).
// ---------------------------------------------------------------------------
template <int N>
__global__ __launch_bounds__(256)
void gate_up_silu_kernel(const __bf16* __restrict__ Xb,
                         const float* __restrict__ WgBase,
                         const float* __restrict__ WuBase,
                         __bf16* __restrict__ Out,
                         const int* __restrict__ tok,
                         const int* __restrict__ counts,
                         const int* __restrict__ offsets,
                         size_t strideW) {
  int e    = blockIdx.z;
  int rows = counts ? counts[e] : T_TOK;
  int m0   = blockIdx.x * TILE_M;
  if (m0 >= rows) return;                 // uniform: whole block exits
  int n0      = blockIdx.y * TILE_N;
  int rowbase = offsets ? offsets[e] : 0;

  __shared__ __align__(16) __bf16 As[TILE_M * LDK];
  __shared__ __align__(16) __bf16 Bg[TILE_N * LDK];
  __shared__ __align__(16) __bf16 Bu[TILE_N * LDK];

  int tid = threadIdx.x, lane = tid & 31, wv = tid >> 5;
  int wm = (wv & 1) * 32;     // wave M offset within tile
  int wn = (wv >> 1) * 32;    // wave N offset within tile

  // A copy: thread -> (row, 16B chunk); gathered token row hoisted (clamped
  // for OOB rows: they feed WMMA but are never stored by the epilogue).
  int arow_l = tid >> 2;
  int acb    = (tid & 3) * 8;
  int agr  = m0 + arow_l;
  int aclm = (agr < rows) ? agr : (rows - 1);
  int atkn = tok ? tok[rowbase + aclm] : aclm;
  const __bf16* a_src = Xb + (size_t)atkn * HDIM + acb;
  __bf16*       a_dst = &As[arow_l * LDK + acb];

  // B staging: thread -> column n (0..127), 16 consecutive k (bkh..bkh+15).
  int bn  = tid & 127;
  int bkh = (tid >> 7) * 16;
  const float* gp = WgBase + (size_t)e * strideW + (size_t)bkh * N + n0 + bn;
  const float* up = WuBase + (size_t)e * strideW + (size_t)bkh * N + n0 + bn;
  __bf16* bg_dst = &Bg[bn * LDK + bkh];
  __bf16* bu_dst = &Bu[bn * LDK + bkh];

  const v8f vzero = {0.f, 0.f, 0.f, 0.f, 0.f, 0.f, 0.f, 0.f};
  v8f cg[2][2], cu[2][2];
  for (int i = 0; i < 2; ++i)
    for (int j = 0; j < 2; ++j) { cg[i][j] = vzero; cu[i][j] = vzero; }

  // pipeline prologue: fetch B regs for iteration 0 (immediate offsets j*N)
  float fg[16], fu[16];
#pragma unroll
  for (int j = 0; j < 16; ++j) {
    fg[j] = gp[(size_t)j * N];
    fu[j] = up[(size_t)j * N];
  }

  const int KITERS = HDIM / TILE_K;
#pragma unroll 1
  for (int it = 0; it < KITERS; ++it) {
    __syncthreads();
    // A tile: async global->LDS (bf16 passthrough, 16B/lane)
    async_cp16(a_dst, a_src);
    a_src += TILE_K;

    // convert staged regs -> packed b128 LDS stores (k-contiguous rows)
    {
      bf16x8 pg0, pg1, pu0, pu1;
#pragma unroll
      for (int j = 0; j < 8; ++j) {
        pg0[j] = (__bf16)fg[j];     pu0[j] = (__bf16)fu[j];
        pg1[j] = (__bf16)fg[j + 8]; pu1[j] = (__bf16)fu[j + 8];
      }
      *reinterpret_cast<bf16x8*>(bg_dst)     = pg0;
      *reinterpret_cast<bf16x8*>(bg_dst + 8) = pg1;
      *reinterpret_cast<bf16x8*>(bu_dst)     = pu0;
      *reinterpret_cast<bf16x8*>(bu_dst + 8) = pu1;
    }

    // fetch B regs for next iteration (latency hidden behind the WMMAs)
    gp += (size_t)TILE_K * N;
    up += (size_t)TILE_K * N;
    if (it + 1 < KITERS) {
#pragma unroll
      for (int j = 0; j < 16; ++j) {
        fg[j] = gp[(size_t)j * N];
        fu[j] = up[(size_t)j * N];
      }
      // nudge the slab after next toward L2
      __builtin_prefetch(gp + (size_t)TILE_K * N, 0, 0);
      __builtin_prefetch(up + (size_t)TILE_K * N, 0, 0);
    }

    async_wait0();
    __syncthreads();

    bf16x16 a0 = ld_frag_a(As, wm,      lane);
    bf16x16 a1 = ld_frag_a(As, wm + 16, lane);
    bf16x16 g0 = ld_frag_b(Bg, wn,      lane);
    bf16x16 g1 = ld_frag_b(Bg, wn + 16, lane);
    bf16x16 u0 = ld_frag_b(Bu, wn,      lane);
    bf16x16 u1 = ld_frag_b(Bu, wn + 16, lane);

    cg[0][0] = wmma_bf16(a0, g0, cg[0][0]);
    cg[0][1] = wmma_bf16(a0, g1, cg[0][1]);
    cg[1][0] = wmma_bf16(a1, g0, cg[1][0]);
    cg[1][1] = wmma_bf16(a1, g1, cg[1][1]);
    cu[0][0] = wmma_bf16(a0, u0, cu[0][0]);
    cu[0][1] = wmma_bf16(a0, u1, cu[0][1]);
    cu[1][0] = wmma_bf16(a1, u0, cu[1][0]);
    cu[1][1] = wmma_bf16(a1, u1, cu[1][1]);
  }

  // ---- epilogue: silu(g)*u -> bf16
  int half = lane >> 4, nl = lane & 15;
  for (int mi = 0; mi < 2; ++mi)
    for (int ni = 0; ni < 2; ++ni) {
      v8f g = cg[mi][ni], u = cu[mi][ni];
      for (int v = 0; v < 8; ++v) {
        int rl = wm + mi * 16 + v + 8 * half;
        int gr = m0 + rl;
        if (gr < rows) {
          float gg = g[v];
          float h  = (gg / (1.f + __expf(-gg))) * u[v];
          int   c  = n0 + wn + ni * 16 + nl;
          Out[(size_t)(rowbase + gr) * N + c] = (__bf16)h;
        }
      }
    }
}

// ---------------------------------------------------------------------------
// Stage 2: Out[token, h] (+)= rowscale * (Hb @ Wd)
//   KSZ is a template constant (1408 / 5632). atomic_mode==0: plain stores
//   (shared pass initializes d_out); ==1: f32 global atomics scaled by wgt.
// ---------------------------------------------------------------------------
template <int KSZ>
__global__ __launch_bounds__(256)
void down_proj_kernel(const __bf16* __restrict__ Hb,
                      const float* __restrict__ WdBase,
                      float* __restrict__ Out,
                      const int* __restrict__ tok, const float* __restrict__ wgt,
                      const int* __restrict__ counts, const int* __restrict__ offsets,
                      const float* __restrict__ shgate,
                      size_t strideW, int atomic_mode) {
  int e    = blockIdx.z;
  int rows = counts ? counts[e] : T_TOK;
  int m0   = blockIdx.x * TILE_M;
  if (m0 >= rows) return;
  int n0      = blockIdx.y * TILE_N;
  int rowbase = offsets ? offsets[e] : 0;

  __shared__ __align__(16) __bf16 As[TILE_M * LDK];
  __shared__ __align__(16) __bf16 Bt[TILE_N * LDK];
  __shared__ int   stok[TILE_M];
  __shared__ float sw[TILE_M];

  int tid = threadIdx.x, lane = tid & 31, wv = tid >> 5;
  int wm = (wv & 1) * 32;
  int wn = (wv >> 1) * 32;

  if (tid < TILE_M) {
    int gr = m0 + tid;
    if (gr < rows) {
      if (tok) { stok[tid] = tok[rowbase + gr]; sw[tid] = wgt[rowbase + gr]; }
      else     { stok[tid] = gr;               sw[tid] = shgate[gr]; }
    } else {
      stok[tid] = -1; sw[tid] = 0.f;
    }
  }

  int arow_l = tid >> 2;
  int acb    = (tid & 3) * 8;
  int agr    = m0 + arow_l;
  int aclm   = (agr < rows) ? agr : (rows - 1);
  const __bf16* a_src = Hb + (size_t)(rowbase + aclm) * KSZ + acb;
  __bf16*       a_dst = &As[arow_l * LDK + acb];

  int bn  = tid & 127;
  int bkh = (tid >> 7) * 16;
  const float* dp = WdBase + (size_t)e * strideW + (size_t)bkh * HDIM + n0 + bn;
  __bf16* bt_dst = &Bt[bn * LDK + bkh];

  const v8f vzero = {0.f, 0.f, 0.f, 0.f, 0.f, 0.f, 0.f, 0.f};
  v8f cd[2][2];
  for (int i = 0; i < 2; ++i)
    for (int j = 0; j < 2; ++j) cd[i][j] = vzero;

  // pipeline prologue
  float fd[16];
#pragma unroll
  for (int j = 0; j < 16; ++j) fd[j] = dp[(size_t)j * HDIM];

  const int KITERS = KSZ / TILE_K;
#pragma unroll 1
  for (int it = 0; it < KITERS; ++it) {
    __syncthreads();
    async_cp16(a_dst, a_src);
    a_src += TILE_K;

    {
      bf16x8 p0, p1;
#pragma unroll
      for (int j = 0; j < 8; ++j) { p0[j] = (__bf16)fd[j]; p1[j] = (__bf16)fd[j + 8]; }
      *reinterpret_cast<bf16x8*>(bt_dst)     = p0;
      *reinterpret_cast<bf16x8*>(bt_dst + 8) = p1;
    }

    dp += (size_t)TILE_K * HDIM;
    if (it + 1 < KITERS) {
#pragma unroll
      for (int j = 0; j < 16; ++j) fd[j] = dp[(size_t)j * HDIM];
      __builtin_prefetch(dp + (size_t)TILE_K * HDIM, 0, 0);
    }

    async_wait0();
    __syncthreads();

    bf16x16 a0 = ld_frag_a(As, wm,      lane);
    bf16x16 a1 = ld_frag_a(As, wm + 16, lane);
    bf16x16 b0 = ld_frag_b(Bt, wn,      lane);
    bf16x16 b1 = ld_frag_b(Bt, wn + 16, lane);

    cd[0][0] = wmma_bf16(a0, b0, cd[0][0]);
    cd[0][1] = wmma_bf16(a0, b1, cd[0][1]);
    cd[1][0] = wmma_bf16(a1, b0, cd[1][0]);
    cd[1][1] = wmma_bf16(a1, b1, cd[1][1]);
  }

  int half = lane >> 4, nl = lane & 15;
  for (int mi = 0; mi < 2; ++mi)
    for (int ni = 0; ni < 2; ++ni) {
      v8f c = cd[mi][ni];
      for (int v = 0; v < 8; ++v) {
        int rl = wm + mi * 16 + v + 8 * half;
        int t  = stok[rl];
        if (t >= 0) {
          float val = c[v] * sw[rl];
          int   col = n0 + wn + ni * 16 + nl;
          float* dst = Out + (size_t)t * HDIM + col;
          if (atomic_mode) unsafeAtomicAdd(dst, val);
          else             *dst = val;
        }
      }
    }
}

// ---------------------------------------------------------------------------
// Host-side launch
// ---------------------------------------------------------------------------
extern "C" void kernel_launch(void* const* d_in, const int* in_sizes, int n_in,
                              void* d_out, int out_size, void* d_ws, size_t ws_size,
                              hipStream_t stream) {
  const float* x       = (const float*)d_in[0];
  const float* rw      = (const float*)d_in[1];
  const float* w_gate  = (const float*)d_in[2];
  const float* w_up    = (const float*)d_in[3];
  const float* w_down  = (const float*)d_in[4];
  const float* sh_gate = (const float*)d_in[5];
  const float* sh_up   = (const float*)d_in[6];
  const float* sh_down = (const float*)d_in[7];
  const float* sh_eg   = (const float*)d_in[8];
  float* out = (float*)d_out;

  // ---- workspace carve (~55 MB total) ----
  char* p = (char*)d_ws;
  auto carve = [&](size_t bytes) {
    void* r = (void*)p;
    p += (bytes + 255) & ~(size_t)255;
    return r;
  };
  __bf16* xb     = (__bf16*)carve((size_t)T_TOK * HDIM * 2);
  __bf16* hbuf   = (__bf16*)carve((size_t)T_TOK * TOPK * FDIM * 2);
  __bf16* sbuf   = (__bf16*)carve((size_t)T_TOK * SFDIM * 2);
  int*    topidx = (int*)   carve((size_t)T_TOK * TOPK * 4);
  float*  topw   = (float*) carve((size_t)T_TOK * TOPK * 4);
  float*  shg    = (float*) carve((size_t)T_TOK * 4);
  int*    counts = (int*)   carve(NEXP * 4);
  int*    curs   = (int*)   carve(NEXP * 4);
  int*    offs   = (int*)   carve((NEXP + 1) * 4);
  int*    tok    = (int*)   carve((size_t)T_TOK * TOPK * 4);
  float*  wgt    = (float*) carve((size_t)T_TOK * TOPK * 4);

  zero_counts_kernel<<<1, 64, 0, stream>>>(counts);
  router_kernel<<<T_TOK, 64, 0, stream>>>(x, rw, sh_eg, topidx, topw, shg, counts);
  prefix_kernel<<<1, 64, 0, stream>>>(counts, offs, curs);
  scatter_kernel<<<T_TOK / 256, 256, 0, stream>>>(topidx, topw, offs, curs, tok, wgt);
  cvt_x_kernel<<<(T_TOK * HDIM) / 1024, 256, 0, stream>>>(x, xb);

  // stage 1: shared expert  (s = silu(x@Wsg)*(x@Wsu), bf16)
  gate_up_silu_kernel<SFDIM>
      <<<dim3(T_TOK / TILE_M, SFDIM / TILE_N, 1), 256, 0, stream>>>(
      xb, sh_gate, sh_up, sbuf, nullptr, nullptr, nullptr, 0);
  // stage 1: sparse experts (h = silu(x@G_e)*(x@U_e) on gathered tokens)
  gate_up_silu_kernel<FDIM>
      <<<dim3(T_TOK / TILE_M, FDIM / TILE_N, NEXP), 256, 0, stream>>>(
      xb, w_gate, w_up, hbuf, tok, counts, offs, (size_t)HDIM * FDIM);

  // stage 2: shared expert down-proj (plain stores -> initializes d_out)
  down_proj_kernel<SFDIM>
      <<<dim3(T_TOK / TILE_M, HDIM / TILE_N, 1), 256, 0, stream>>>(
      sbuf, sh_down, out, nullptr, nullptr, nullptr, nullptr, shg,
      0, /*atomic=*/0);
  // stage 2: sparse experts down-proj (scaled, atomic accumulate)
  down_proj_kernel<FDIM>
      <<<dim3(T_TOK / TILE_M, HDIM / TILE_N, NEXP), 256, 0, stream>>>(
      hbuf, w_down, out, tok, wgt, counts, offs, nullptr,
      (size_t)FDIM * HDIM, /*atomic=*/1);
  (void)in_sizes; (void)n_in; (void)out_size; (void)ws_size;
}
